// GateV2_89163521065174
// MI455X (gfx1250) — compile-verified, matrix operand order checked by reading, not run on hardware
//
#include <hip/hip_runtime.h>

#define NODES   10000
#define DFEAT   128
#define ETOT    320000
#define NBLK    2500      // ETOT / 128 edges per block
#define KTILES  12        // 384 / 32
#define NTILES  8         // 128 / 16

typedef __attribute__((ext_vector_type(16))) __bf16          v16bf;
typedef __attribute__((ext_vector_type(8)))  float           v8f;
typedef __attribute__((ext_vector_type(8)))  unsigned int    v8u;
typedef __attribute__((ext_vector_type(4)))  float           v4f;

__global__ void zero_out_kernel(float* __restrict__ out, int n) {
  int i = blockIdx.x * blockDim.x + threadIdx.x;
  if (i < n) out[i] = 0.0f;
}

// Swizzle W1 (384x128 f32, row-major / K-major) into bf16 WMMA B-fragment order.
// ws element index = ((kt*8 + nt)*32 + lane)*16 + j ; per-lane 16 bf16 = one B frag.
// Lane holds column n = nt*16 + lane%16 ; half = lane/16 ;
// j<8  -> K = kt*32 + half*8 + j ; j>=8 -> K = kt*32 + 16 + half*8 + (j-8).
__global__ void prep_w1_kernel(const float* __restrict__ W1,
                               unsigned short* __restrict__ wsB) {
  int tid = blockIdx.x * blockDim.x + threadIdx.x;
  if (tid >= KTILES * NTILES * 32 * 16) return;
  int j    = tid & 15;
  int lane = (tid >> 4) & 31;
  int nt   = (tid >> 9) & 7;
  int kt   = tid >> 12;
  int half = lane >> 4;
  int n = nt * 16 + (lane & 15);
  int k = kt * 32 + ((j < 8) ? (half * 8 + j) : (16 + half * 8 + (j - 8)));
  ((__bf16*)wsB)[tid] = (__bf16)W1[k * 128 + n];
}

__device__ __forceinline__ v16bf make_afrag(v4f a0, v4f a1, v4f a2, v4f a3) {
  v16bf r;
  r[0]  = (__bf16)a0[0]; r[1]  = (__bf16)a0[1]; r[2]  = (__bf16)a0[2]; r[3]  = (__bf16)a0[3];
  r[4]  = (__bf16)a1[0]; r[5]  = (__bf16)a1[1]; r[6]  = (__bf16)a1[2]; r[7]  = (__bf16)a1[3];
  r[8]  = (__bf16)a2[0]; r[9]  = (__bf16)a2[1]; r[10] = (__bf16)a2[2]; r[11] = (__bf16)a2[3];
  r[12] = (__bf16)a3[0]; r[13] = (__bf16)a3[1]; r[14] = (__bf16)a3[2]; r[15] = (__bf16)a3[3];
  return r;
}

__launch_bounds__(256)
__global__ void gate_main_kernel(const float* __restrict__ msg,
                                 const float* __restrict__ x_i,
                                 const float* __restrict__ x_j,
                                 const float* __restrict__ e_ij,
                                 const unsigned short* __restrict__ wsB,
                                 const float* __restrict__ b1,
                                 const float* __restrict__ W2,
                                 const float* __restrict__ b2,
                                 const int* __restrict__ index,
                                 float* __restrict__ out) {
  __shared__ float w_lds[8][16];

  const int tid  = threadIdx.x;
  const int wave = tid >> 5;
  const int lane = tid & 31;
  const int half = lane >> 4;
  const int col  = lane & 15;

  // Per-lane slices of W2/b1: column n = nt*16 + col (matches C-matrix N=lane%16).
  float w2v[NTILES], b1v[NTILES];
#pragma unroll
  for (int nt = 0; nt < NTILES; ++nt) {
    w2v[nt] = W2[nt * 16 + col];
    b1v[nt] = b1[nt * 16 + col];
  }
  const float b2s = b2[0];

  for (int blk = blockIdx.x; blk < NBLK; blk += gridDim.x) {
    const int eBase = blk * 128 + wave * 16;   // this wave's 16 edges

    v8f acc[NTILES] = {};                      // 16x128 f32 accumulators

    const float* const srcs[3] = {x_j, e_ij, x_i};
#pragma unroll
    for (int seg = 0; seg < 3; ++seg) {
      const float* __restrict__ src = srcs[seg];
      // per-lane A row base for this segment
      const float* rowp = src + (size_t)(eBase + col) * DFEAT + half * 8;
#pragma unroll 1
      for (int kt4 = 0; kt4 < 4; ++kt4) {
        const int ks = seg * 4 + kt4;           // global K tile 0..11
        const float* p = rowp + kt4 * 32;

        // A fragment: 2x 32B contiguous f32 segments per lane, streamed (read-once)
        v4f a0 = __builtin_nontemporal_load((const v4f*)(p));
        v4f a1 = __builtin_nontemporal_load((const v4f*)(p + 4));
        v4f a2 = __builtin_nontemporal_load((const v4f*)(p + 16));
        v4f a3 = __builtin_nontemporal_load((const v4f*)(p + 20));
        v16bf aF = make_afrag(a0, a1, a2, a3);

        // B fragments: pre-swizzled, L2-resident; 2-deep software pipeline
        const v8u* bbase = (const v8u*)wsB + (size_t)ks * NTILES * 32 + lane;
        v8u braw = bbase[0];
#pragma unroll
        for (int nt = 0; nt < NTILES; ++nt) {
          v8u nextb = braw;
          if (nt < NTILES - 1) nextb = bbase[(nt + 1) * 32];
          acc[nt] = __builtin_amdgcn_wmma_f32_16x16x32_bf16(
              false, aF, false, __builtin_bit_cast(v16bf, braw),
              (short)0, acc[nt], false, false);
          braw = nextb;
        }
      }
    }

    // Epilogue: bias + leaky_relu + dot(W2) + tanh.
    // C layout: VGPR v, lane L -> M = v + 8*(L/16), N = L%16.
#pragma unroll
    for (int v = 0; v < 8; ++v) {
      float s = 0.0f;
#pragma unroll
      for (int nt = 0; nt < NTILES; ++nt) {
        float h = acc[nt][v] + b1v[nt];
        h = (h > 0.0f) ? h : 0.01f * h;   // leaky_relu(0.01)
        s += h * w2v[nt];
      }
      // reduce over the 16 lanes holding this row's 16-column slices
      s += __shfl_xor(s, 1, 32);
      s += __shfl_xor(s, 2, 32);
      s += __shfl_xor(s, 4, 32);
      s += __shfl_xor(s, 8, 32);
      if (col == 0) w_lds[wave][half * 8 + v] = tanhf(s + b2s);
    }
    __syncthreads();

    // Scatter: out[index[e]] += w[e] * msg[e]; lane handles 4 contiguous floats.
#pragma unroll 4
    for (int r = 0; r < 16; ++r) {
      const int e = eBase + r;
      const float w = w_lds[wave][r];
      const v4f mv = __builtin_nontemporal_load(
          (const v4f*)(msg + (size_t)e * DFEAT + lane * 4));
      float* dst = out + (size_t)index[e] * DFEAT + lane * 4;
      unsafeAtomicAdd(dst + 0, w * mv[0]);
      unsafeAtomicAdd(dst + 1, w * mv[1]);
      unsafeAtomicAdd(dst + 2, w * mv[2]);
      unsafeAtomicAdd(dst + 3, w * mv[3]);
    }
    __syncthreads();
  }
}

extern "C" void kernel_launch(void* const* d_in, const int* in_sizes, int n_in,
                              void* d_out, int out_size, void* d_ws, size_t ws_size,
                              hipStream_t stream) {
  const float* msg  = (const float*)d_in[0];
  const float* x_i  = (const float*)d_in[1];
  const float* x_j  = (const float*)d_in[2];
  const float* e_ij = (const float*)d_in[3];
  const float* W1   = (const float*)d_in[4];
  const float* b1   = (const float*)d_in[5];
  const float* W2   = (const float*)d_in[6];
  const float* b2   = (const float*)d_in[7];
  const int*   idx  = (const int*)d_in[8];
  float* out = (float*)d_out;
  unsigned short* wsB = (unsigned short*)d_ws;   // 384*128 bf16 = 96 KB

  const int outElems = NODES * DFEAT;
  zero_out_kernel<<<(outElems + 255) / 256, 256, 0, stream>>>(out, outElems);

  const int wElems = KTILES * NTILES * 32 * 16;  // 49152
  prep_w1_kernel<<<(wElems + 255) / 256, 256, 0, stream>>>(W1, wsB);

  gate_main_kernel<<<NBLK, 256, 0, stream>>>(msg, x_i, x_j, e_ij, wsB,
                                             b1, W2, b2, idx, out);
}
